// DeformableGConvBlock_58634893525709
// MI455X (gfx1250) — matrix operand order repaired and checked by reading, not compiled, not run on hardware
//
#include <hip/hip_runtime.h>
#include <hip/hip_bf16.h>
#include <math.h>

#define N_NODES 16384
#define K_NBR   16
#define N_HOPS  4
#define IN_DIM  256
#define RAW_DIM 128
#define OUT_DIM 64
#define PHI_DIM 16
#define N_KERN  4
#define E_EDGES (N_NODES * K_NBR)

typedef __attribute__((ext_vector_type(16))) _Float16 v16h;
typedef __attribute__((ext_vector_type(8)))  float    v8f;

// ---------------------------------------------------------------------------
// Fragment-packing layouts (wave32 WMMA f16 16x16x32, per CDNA5 ISA 7.12.2):
//  A (16x32, MxK): lane = (m&15) + 16*hi, hi selects K half;
//                  element j -> K = 32*ktile + 16*(j>>3) + 8*hi + (j&7)
//  B (32x16, KxN): lane = (n&15) + 16*hi, hi selects K half;
//                  element j -> K = 32*ktile + 16*(j>>3) + 8*hi + (j&7)
// Packed arrays are stored so each lane's 16 halfs (32B) are contiguous.
// ---------------------------------------------------------------------------

__global__ void pack_a_f32(const float* __restrict__ A, _Float16* __restrict__ out,
                           int M, int K) {
  int idx = blockIdx.x * blockDim.x + threadIdx.x;
  if (idx >= M * K) return;
  int m = idx / K, k = idx % K;
  int kt = k >> 5, kk = k & 31;
  int g = kk >> 4, rem = kk & 15, hi = rem >> 3, jj = rem & 7;
  int j = 8 * g + jj;
  int lane = (m & 15) + 16 * hi;
  out[(((size_t)((m >> 4) * (K >> 5) + kt)) * 32 + lane) * 16 + j] = (_Float16)A[idx];
}

__global__ void pack_b_f32(const float* __restrict__ B, _Float16* __restrict__ out,
                           int K, int Nin, int Npad) {
  int idx = blockIdx.x * blockDim.x + threadIdx.x;
  if (idx >= K * Npad) return;
  int k = idx / Npad, n = idx % Npad;
  float v = (n < Nin) ? B[(size_t)k * Nin + n] : 0.0f;
  int kt = k >> 5, kk = k & 31;
  int lane = (n & 15) + 16 * ((kk >> 3) & 1);
  int j = 8 * (kk >> 4) + (kk & 7);
  out[(((size_t)(kt * (Npad >> 4) + (n >> 4))) * 32 + lane) * 16 + j] = (_Float16)v;
}

// ---------------------------------------------------------------------------
// WMMA GEMM: C[M,N] = A[M,K] @ B[K,N] (+bias) (+relu), K = 32*KTILES fixed at
// compile time. Each wave owns a 16 x (16*NT) tile; blockDim=256 (8 waves).
// Software-pipelined + fully unrolled: prefetch of k-tile kt+1 overlaps the
// WMMAs of tile kt, and full unroll turns the double-buffer rotation into SSA
// renames (no v_mov copies).
// MODE 0: store f32 row-major. MODE 1: store f16 in A-packed layout (feeds the
// next GEMM directly, fusing relu + repack).
// ---------------------------------------------------------------------------
template <int NT, int MODE, int KTILES>
__global__ __launch_bounds__(256) void gemm_wmma_kernel(
    const v16h* __restrict__ Ap, const v16h* __restrict__ Bp,
    const float* __restrict__ bias, int has_bias, int relu,
    float* __restrict__ Cf32, _Float16* __restrict__ Cpack, int N) {
  const int lane   = threadIdx.x & 31;
  const int wv     = threadIdx.x >> 5;
  const int mtile  = blockIdx.x * 8 + wv;
  const int ntile0 = blockIdx.y * NT;
  const int ntiles = N >> 4;

  v8f c[NT];
#pragma unroll
  for (int t = 0; t < NT; ++t) { v8f zz = {}; c[t] = zz; }

  const v16h* Abase = Ap + (size_t)mtile * KTILES * 32 + lane;
  const v16h* Bbase = Bp + (size_t)ntile0 * 32 + lane;

  // prologue: fragments for k-tile 0
  v16h a_cur = Abase[0];
  v16h b_cur[NT];
#pragma unroll
  for (int t = 0; t < NT; ++t) b_cur[t] = Bbase[(size_t)t * 32];

#pragma unroll
  for (int kt = 0; kt < KTILES - 1; ++kt) {
    // prefetch k-tile kt+1 while WMMAs for kt execute
    v16h a_nxt = Abase[(size_t)(kt + 1) * 32];
    v16h b_nxt[NT];
#pragma unroll
    for (int t = 0; t < NT; ++t)
      b_nxt[t] = Bbase[((size_t)(kt + 1) * ntiles + t) * 32];
#pragma unroll
    for (int t = 0; t < NT; ++t)
      c[t] = __builtin_amdgcn_wmma_f32_16x16x32_f16(false, a_cur, false, b_cur[t],
                                                    (short)0, c[t], false, false);
    a_cur = a_nxt;
#pragma unroll
    for (int t = 0; t < NT; ++t) b_cur[t] = b_nxt[t];
  }
  // epilogue: last k-tile
#pragma unroll
  for (int t = 0; t < NT; ++t)
    c[t] = __builtin_amdgcn_wmma_f32_16x16x32_f16(false, a_cur, false, b_cur[t],
                                                  (short)0, c[t], false, false);

  const int hi = lane >> 4;
  const int nc = lane & 15;
#pragma unroll
  for (int t = 0; t < NT; ++t) {
    const int col = (ntile0 + t) * 16 + nc;
    const float bv = has_bias ? bias[col] : 0.0f;
#pragma unroll
    for (int r = 0; r < 8; ++r) {
      const int m = mtile * 16 + r + 8 * hi;
      float v = c[t][r] + bv;
      if (relu) v = fmaxf(v, 0.0f);
      if (MODE == 0) {
        Cf32[(size_t)m * N + col] = v;
      } else {
        const int kk = col & 31, kt2 = col >> 5;
        const int g = kk >> 4, rem = kk & 15, hi2 = rem >> 3, jj = rem & 7;
        const int j = 8 * g + jj;
        const int lane2 = (m & 15) + 16 * hi2;
        Cpack[(((size_t)mtile * (N >> 5) + kt2) * 32 + lane2) * 16 + j] = (_Float16)v;
      }
    }
  }
}

// ---------------------------------------------------------------------------
// tn = normalize(tilde_phi rows) for all hops; l_sep = mean_h [sum_ij ||ti-tj||^2 / nk]
// Single block of 64 threads.
// ---------------------------------------------------------------------------
__global__ void tn_lsep_kernel(const float* __restrict__ tphi, float* __restrict__ tn_ws,
                               float* __restrict__ lsep_out) {
  __shared__ float s_tn[N_HOPS * N_KERN * PHI_DIM];
  __shared__ float red[64];
  int tid = threadIdx.x;
  if (tid < N_HOPS * N_KERN) {
    const float* r = tphi + tid * PHI_DIM;
    float nn = 0.0f;
    for (int p = 0; p < PHI_DIM; ++p) nn += r[p] * r[p];
    float inv = 1.0f / sqrtf(nn);
    for (int p = 0; p < PHI_DIM; ++p) {
      float v = r[p] * inv;
      s_tn[tid * PHI_DIM + p] = v;
      tn_ws[tid * PHI_DIM + p] = v;
    }
  }
  __syncthreads();
  int h = tid >> 4, i = (tid >> 2) & 3, j = tid & 3;
  float acc = 0.0f;
  for (int p = 0; p < PHI_DIM; ++p) {
    float d = s_tn[(h * 4 + i) * PHI_DIM + p] - s_tn[(h * 4 + j) * PHI_DIM + p];
    acc += d * d;
  }
  red[tid] = acc;
  __syncthreads();
  for (int s = 32; s > 0; s >>= 1) {
    if (tid < s) red[tid] += red[tid + s];
    __syncthreads();
  }
  if (tid == 0) *lsep_out = red[0] * (1.0f / (N_KERN * N_HOPS));
}

// delta (N x 64) -> delta_ik = delta + tn[k][p] (in place); accumulate l_focus.
__global__ void deltafix_kernel(float* __restrict__ delta, const float* __restrict__ tn,
                                float* __restrict__ lfocus, float scale) {
  __shared__ float red[256];
  int idx = blockIdx.x * 256 + threadIdx.x;
  float v = 0.0f;
  if (idx < N_NODES * 64) {
    float d = delta[idx];
    v = d * d;
    delta[idx] = d + tn[idx & 63];
  }
  red[threadIdx.x] = v;
  __syncthreads();
  for (int s = 128; s > 0; s >>= 1) {
    if (threadIdx.x < s) red[threadIdx.x] += red[threadIdx.x + s];
    __syncthreads();
  }
  if (threadIdx.x == 0) atomicAdd(lfocus, red[0] * scale);
}

__global__ void zero_scalar_kernel(float* p) { if (threadIdx.x == 0) *p = 0.0f; }

// ---------------------------------------------------------------------------
// Edge stage: one wave per dst node (its 16 incoming edges are contiguous).
// Phase A (lanes 0..15 = edges): dist, normalize, logits, 16-wide softmax.
// Phase B (32 lanes x 2 cols): attention-weighted gather-sum of hk[src] (L2 hits),
// bias add, row-normalize.
// ---------------------------------------------------------------------------
__global__ __launch_bounds__(256) void edge_agg_kernel(
    const float* __restrict__ hk, const float* __restrict__ phi,
    const float* __restrict__ dik, const int* __restrict__ src,
    const float* __restrict__ bias, float* __restrict__ outl) {
  __shared__ float s_att[8][64];
  __shared__ int s_src[8][16];
  const int lane = threadIdx.x & 31;
  const int wv = threadIdx.x >> 5;
  const int n = blockIdx.x * 8 + wv;

  if (lane < 16) {
    int s = src[n * K_NBR + lane];
    s_src[wv][lane] = s;
    const float* ps = phi + (size_t)s * PHI_DIM;
    const float* pn = phi + (size_t)n * PHI_DIM;
    float d[PHI_DIM - 1];
    float nrm = 0.0f;
    int allz = 1;
#pragma unroll
    for (int p = 0; p < PHI_DIM - 1; ++p) {
      float dv = ps[p] - pn[p];
      d[p] = dv;
      nrm += dv * dv;
      allz &= (dv == 0.0f);
    }
    float last = allz ? 1.0f : 0.0f;
    nrm += last;  // last^2 == last
    float rn = 1.0f / fmaxf(sqrtf(nrm), 1e-8f);
    float lg[N_KERN];
#pragma unroll
    for (int k = 0; k < N_KERN; ++k) {
      const float* dk = dik + (size_t)n * 64 + k * PHI_DIM;
      float acc = last * dk[PHI_DIM - 1];
#pragma unroll
      for (int p = 0; p < PHI_DIM - 1; ++p) acc += d[p] * dk[p];
      lg[k] = acc * rn;
    }
#pragma unroll
    for (int k = 0; k < N_KERN; ++k) {
      float mx = lg[k];
      for (int m = 1; m < 16; m <<= 1) mx = fmaxf(mx, __shfl_xor(mx, m, 16));
      float ex = __expf(lg[k] - mx);
      float sm = ex;
      for (int m = 1; m < 16; m <<= 1) sm += __shfl_xor(sm, m, 16);
      s_att[wv][k * 16 + lane] = ex / sm;
    }
  }
  __syncthreads();

  float2 acc = make_float2(0.0f, 0.0f);
#pragma unroll 4
  for (int e = 0; e < K_NBR; ++e) {
    int s = s_src[wv][e];
    const float* base = hk + (size_t)s * (N_KERN * OUT_DIM);
#pragma unroll
    for (int k = 0; k < N_KERN; ++k) {
      float a = s_att[wv][k * 16 + e];
      float2 hv = ((const float2*)(base + k * OUT_DIM))[lane];
      acc.x += a * hv.x;
      acc.y += a * hv.y;
    }
  }
  acc.x += bias[2 * lane];
  acc.y += bias[2 * lane + 1];
  float nn = acc.x * acc.x + acc.y * acc.y;
  for (int m = 1; m < 32; m <<= 1) nn += __shfl_xor(nn, m, 32);
  float r = 1.0f / fmaxf(sqrtf(nn), 1e-8f);
  float* o = outl + (size_t)n * OUT_DIM + 2 * lane;
  o[0] = acc.x * r;
  o[1] = acc.y * r;
}

// Hop-softmax fusion: one wave per node, 2 cols per lane.
__global__ __launch_bounds__(256) void fuse_kernel(const float* __restrict__ outs,
                                                   const float* __restrict__ z,
                                                   float* __restrict__ th) {
  const int lane = threadIdx.x & 31;
  const int wv = threadIdx.x >> 5;
  const int n = blockIdx.x * 8 + wv;
  float zx = z[2 * lane], zy = z[2 * lane + 1];
  float2 o[N_HOPS];
  float sc[N_HOPS];
#pragma unroll
  for (int l = 0; l < N_HOPS; ++l) {
    o[l] = ((const float2*)(outs + ((size_t)l * N_NODES + n) * OUT_DIM))[lane];
    float p = o[l].x * zx + o[l].y * zy;
    for (int m = 1; m < 32; m <<= 1) p += __shfl_xor(p, m, 32);
    sc[l] = p;
  }
  float mx = fmaxf(fmaxf(sc[0], sc[1]), fmaxf(sc[2], sc[3]));
  float ex[N_HOPS], sm = 0.0f;
#pragma unroll
  for (int l = 0; l < N_HOPS; ++l) { ex[l] = __expf(sc[l] - mx); sm += ex[l]; }
  float2 acc = make_float2(0.0f, 0.0f);
#pragma unroll
  for (int l = 0; l < N_HOPS; ++l) {
    float a = ex[l] / sm;
    acc.x += a * o[l].x;
    acc.y += a * o[l].y;
  }
  float* d = th + (size_t)n * OUT_DIM + 2 * lane;
  d[0] = acc.x;
  d[1] = acc.y;
}

// ---------------------------------------------------------------------------
extern "C" void kernel_launch(void* const* d_in, const int* in_sizes, int n_in,
                              void* d_out, int out_size, void* d_ws, size_t ws_size,
                              hipStream_t stream) {
  (void)in_sizes; (void)n_in; (void)out_size; (void)ws_size;
  const float* h        = (const float*)d_in[0];
  const float* e_tensor = (const float*)d_in[1];
  const int*   src      = (const int*)d_in[2];
  /* d_in[3] = dst: structurally repeat(arange(N),16); edges contiguous per dst */
  const float* W_phi = (const float*)d_in[4];
  const float* w1    = (const float*)d_in[5];
  const float* b1    = (const float*)d_in[6];
  const float* w2    = (const float*)d_in[7];
  const float* b2    = (const float*)d_in[8];
  const float* tphi  = (const float*)d_in[9];
  const float* Wmat  = (const float*)d_in[10];
  const float* bias  = (const float*)d_in[11];
  const float* z     = (const float*)d_in[12];

  float* th     = (float*)d_out;
  float* lsep   = th + (size_t)N_NODES * OUT_DIM;
  float* lfocus = lsep + 1;

  char* ws = (char*)d_ws;
  size_t off = 0;
  auto carve = [&](size_t bytes) -> void* {
    void* p = ws + off;
    off = (off + bytes + 255) & ~(size_t)255;
    return p;
  };
  _Float16* h_pack   = (_Float16*)carve((size_t)N_NODES * IN_DIM * 2);
  _Float16* e_pack   = (_Float16*)carve((size_t)N_NODES * RAW_DIM * 2);
  _Float16* hid_pack = (_Float16*)carve((size_t)N_NODES * IN_DIM * 2);
  _Float16* w1p      = (_Float16*)carve((size_t)RAW_DIM * IN_DIM * 2);
  _Float16* w2p      = (_Float16*)carve((size_t)IN_DIM * 64 * 2);
  _Float16* wphip    = (_Float16*)carve((size_t)RAW_DIM * PHI_DIM * 2);
  _Float16* Wp       = (_Float16*)carve((size_t)IN_DIM * 256 * 2);
  float* delta = (float*)carve((size_t)N_NODES * 64 * 4);
  float* phi   = (float*)carve((size_t)N_NODES * PHI_DIM * 4);
  float* hk    = (float*)carve((size_t)N_NODES * 256 * 4);
  float* outs  = (float*)carve((size_t)N_HOPS * N_NODES * OUT_DIM * 4);
  float* tn_ws = (float*)carve((size_t)N_HOPS * N_KERN * PHI_DIM * 4);

  // scalars: l_focus accumulated via atomics -> zero each call; l_sep stored.
  zero_scalar_kernel<<<1, 32, 0, stream>>>(lfocus);
  tn_lsep_kernel<<<1, 64, 0, stream>>>(tphi, tn_ws, lsep);

  // pack h once (shared by all hops)
  pack_a_f32<<<(N_NODES * IN_DIM + 255) / 256, 256, 0, stream>>>(h, h_pack, N_NODES, IN_DIM);

  const int MT_BLOCKS = N_NODES / 16 / 8;  // 128: 8 M-tiles per block

  for (int l = 0; l < N_HOPS; ++l) {
    const float* e_l = e_tensor + (size_t)l * N_NODES * RAW_DIM;
    pack_a_f32<<<(N_NODES * RAW_DIM + 255) / 256, 256, 0, stream>>>(e_l, e_pack, N_NODES, RAW_DIM);
    pack_b_f32<<<(RAW_DIM * IN_DIM + 255) / 256, 256, 0, stream>>>(
        w1 + (size_t)l * RAW_DIM * IN_DIM, w1p, RAW_DIM, IN_DIM, IN_DIM);
    pack_b_f32<<<(IN_DIM * 64 + 255) / 256, 256, 0, stream>>>(
        w2 + (size_t)l * IN_DIM * 64, w2p, IN_DIM, 64, 64);
    pack_b_f32<<<(RAW_DIM * PHI_DIM + 255) / 256, 256, 0, stream>>>(
        W_phi + (size_t)l * RAW_DIM * (PHI_DIM - 1), wphip, RAW_DIM, PHI_DIM - 1, PHI_DIM);
    pack_b_f32<<<(IN_DIM * 256 + 255) / 256, 256, 0, stream>>>(
        Wmat + (size_t)l * IN_DIM * 256, Wp, IN_DIM, 256, 256);

    // hid = relu(e @ w1 + b1), written directly in A-packed f16 layout (K=128 -> 4 k-tiles)
    gemm_wmma_kernel<4, 1, RAW_DIM / 32><<<dim3(MT_BLOCKS, 4), 256, 0, stream>>>(
        (const v16h*)e_pack, (const v16h*)w1p, b1 + (size_t)l * IN_DIM, 1, 1,
        nullptr, hid_pack, IN_DIM);
    // delta = hid @ w2 + b2 (K=256 -> 8 k-tiles)
    gemm_wmma_kernel<4, 0, IN_DIM / 32><<<dim3(MT_BLOCKS, 1), 256, 0, stream>>>(
        (const v16h*)hid_pack, (const v16h*)w2p, b2 + (size_t)l * 64, 1, 0,
        delta, nullptr, 64);
    // delta_ik = tn + delta (in place) + l_focus accumulation
    deltafix_kernel<<<(N_NODES * 64) / 256, 256, 0, stream>>>(
        delta, tn_ws + (size_t)l * N_KERN * PHI_DIM, lfocus,
        1.0f / ((float)N_NODES * N_KERN * N_HOPS));
    // phi = e @ W_phi (padded to 16 cols, col15 = 0; K=128 -> 4 k-tiles)
    gemm_wmma_kernel<1, 0, RAW_DIM / 32><<<dim3(MT_BLOCKS, 1), 256, 0, stream>>>(
        (const v16h*)e_pack, (const v16h*)wphip, nullptr, 0, 0,
        phi, nullptr, PHI_DIM);
    // hk = h @ W (K=256 -> 8 k-tiles)
    gemm_wmma_kernel<4, 0, IN_DIM / 32><<<dim3(MT_BLOCKS, 4), 256, 0, stream>>>(
        (const v16h*)h_pack, (const v16h*)Wp, nullptr, 0, 0,
        hk, nullptr, 256);
    // edge softmax + message passing + row-normalize
    edge_agg_kernel<<<N_NODES / 8, 256, 0, stream>>>(
        hk, phi, delta, src + (size_t)l * E_EDGES, bias + (size_t)l * OUT_DIM,
        outs + (size_t)l * N_NODES * OUT_DIM);
  }

  // hop-level softmax fusion
  fuse_kernel<<<N_NODES / 8, 256, 0, stream>>>(outs, z, th);
}
